// DPASSMBlock_56057913147960
// MI455X (gfx1250) — matrix operand order repaired and checked by reading, not compiled, not run on hardware
//
#include <hip/hip_runtime.h>

// ---------------------------------------------------------------------------
// CDNA5 (gfx1250) wave32 + WMMA bf16 implementation of the DPA-SSM block.
// ---------------------------------------------------------------------------

typedef __attribute__((ext_vector_type(16))) __bf16 v16bf;
typedef __attribute__((ext_vector_type(8)))  float  v8f;

union U16 { unsigned u[8]; v16bf v; };

__device__ __forceinline__ unsigned short f2bf(float f) {
  unsigned u = __float_as_uint(f);
  u += 0x7FFFu + ((u >> 16) & 1u);          // round-to-nearest-even
  return (unsigned short)(u >> 16);
}

__device__ __forceinline__ v8f v8f_zero() {
  v8f z;
#pragma unroll
  for (int i = 0; i < 8; i++) z[i] = 0.f;
  return z;
}

__device__ __forceinline__ v8f wmma_bf16(const v16bf& a, const v16bf& b, v8f c) {
  return __builtin_amdgcn_wmma_f32_16x16x32_bf16(false, a, false, b, (short)0, c,
                                                 false, false);
}

// CDNA5 async global->LDS copy (ASYNCcnt-tracked, no VGPR staging).
__device__ __forceinline__ void async_load_b128(unsigned lds_off,
                                                const void* gaddr) {
  asm volatile("global_load_async_to_lds_b128 %0, %1, off"
               :
               : "v"(lds_off), "v"((unsigned long long)(size_t)gaddr)
               : "memory");
}
__device__ __forceinline__ void wait_async0() {
  asm volatile("s_wait_asynccnt 0" ::: "memory");
}

// ---------------------------------------------------------------------------
// Generic GEMM: C[M,N] = A[M,K](bf16) * B[K,N](bf16) (+bias)(+GELU)(+resid)
// 256 threads = 8 waves; block tile 128x128; K-step 32.
// A tile staged via global_load_async_to_lds_b128; B tile transposed 2x2 in
// registers so all global/LDS traffic is >=32-bit.
// ---------------------------------------------------------------------------
#define TBM 128
#define TBN 128
#define TBK 32

__global__ __launch_bounds__(256) void gemm_bf16_wmma(
    const unsigned short* __restrict__ A, const unsigned short* __restrict__ B,
    const float* __restrict__ bias, const float* __restrict__ resid,
    float* __restrict__ Cf, unsigned short* __restrict__ Cb,
    int M, int N, int K, int act) {
  __shared__ unsigned As[TBM * TBK / 2];      // [128][16] u32 (k-pairs, row-major)
  __shared__ unsigned Bs[TBN * TBK / 2];      // [128][16] u32 (n-major, k-pairs)

  const unsigned* Bu = (const unsigned*)B;
  const int tid = threadIdx.x;
  const int w = tid >> 5, lane = tid & 31, ln = lane & 15, hi = lane >> 4;
  const int rowBase = blockIdx.y * TBM, colBase = blockIdx.x * TBN;
  const int Nu = N >> 1;
  const int colBase2 = colBase >> 1;

  v8f acc[8];
#pragma unroll
  for (int i = 0; i < 8; i++) acc[i] = v8f_zero();

  for (int k0 = 0; k0 < K; k0 += TBK) {
    __syncthreads();                          // previous compute done with LDS

    // A tile: 128 rows x 64B -> 512 B128 async transfers (2 per thread)
#pragma unroll
    for (int i = 0; i < 2; i++) {
      int idx = tid + i * 256;
      int r = idx >> 2, q = idx & 3;
      const char* g = (const char*)A +
                      (((size_t)(rowBase + r) * K + k0) << 1) + q * 16;
      async_load_b128((unsigned)(size_t)(const void*)&As[idx * 4], g);
    }

    // B tile: 32x128 halfs, transposed via 2x2 blocks (u32 in, u32 out)
#pragma unroll
    for (int i = 0; i < 4; i++) {
      int idx = tid + i * 256;                // 1024 2x2 blocks
      int np = idx & 63, kp = idx >> 6;
      unsigned a = Bu[(size_t)(k0 + 2 * kp) * Nu + colBase2 + np];
      unsigned b = Bu[(size_t)(k0 + 2 * kp + 1) * Nu + colBase2 + np];
      Bs[(np * 2) * 16 + kp]     = (a & 0xFFFFu) | (b << 16);
      Bs[(np * 2 + 1) * 16 + kp] = (a >> 16) | (b & 0xFFFF0000u);
    }

    wait_async0();                            // drain ASYNCcnt before barrier
    __syncthreads();

    // prefetch next K-step's tiles into caches
    if (k0 + TBK < K) {
      __builtin_prefetch((const char*)A +
                         (((size_t)(rowBase + (tid >> 1)) * K + k0 + TBK) << 1));
      __builtin_prefetch((const char*)B +
                         (((size_t)(k0 + TBK + (tid >> 6)) * N + colBase) << 1));
    }

    U16 a;
    const int arow = w * 16 + ln;
#pragma unroll
    for (int i = 0; i < 4; i++) {             // A fragment: 2 contiguous chunks
      a.u[i]     = As[arow * 16 + hi * 4 + i];
      a.u[4 + i] = As[arow * 16 + 8 + hi * 4 + i];
    }
#pragma unroll
    for (int nt = 0; nt < 8; nt++) {
      U16 b;
      const int bn = nt * 16 + ln;
#pragma unroll
      for (int j = 0; j < 8; j++) b.u[j] = Bs[bn * 16 + (j >> 2) * 8 + hi * 4 + (j & 3)];
      acc[nt] = wmma_bf16(a.v, b.v, acc[nt]);
    }
  }

#pragma unroll
  for (int nt = 0; nt < 8; nt++) {
    const int col = colBase + nt * 16 + ln;
    const float bv = bias ? bias[col] : 0.f;
#pragma unroll
    for (int j = 0; j < 8; j++) {
      const int row = rowBase + w * 16 + hi * 8 + j;
      float v = acc[nt][j] + bv;
      if (act == 1) v = 0.5f * v * (1.f + erff(v * 0.70710678118654752f));
      if (resid) v += resid[(size_t)row * N + col];
      if (Cf) Cf[(size_t)row * N + col] = v;
      if (Cb) Cb[(size_t)row * N + col] = f2bf(v);
    }
  }
}

// ---------------------------------------------------------------------------
// Flash-style windowed attention. grid = (16 heads, 16 blocks of 256 queries).
// Each wave owns 32 query rows; online softmax; P bounced via per-wave LDS.
// qkv: bf16 [4096][3072] (q | k | v).  o: bf16 [4096][1024].
// ---------------------------------------------------------------------------
__global__ __launch_bounds__(256) void attn_wmma(
    const unsigned short* __restrict__ qkv, unsigned short* __restrict__ o) {
  __shared__ unsigned short Ks[512 * 64];     // K window, natural [key][dim]
  __shared__ unsigned short VT[64 * 514];     // V transposed [dim][key], padded
  __shared__ unsigned short Pb[8 * 32 * 32];  // per-wave P bounce tiles

  const int h = blockIdx.x, blk = blockIdx.y;
  const int tid = threadIdx.x, w = tid >> 5, lane = tid & 31;
  const int ln = lane & 15, hi = lane >> 4;
  const unsigned* qkvu = (const unsigned*)qkv;
  unsigned* Ksw = (unsigned*)Ks;

#pragma unroll 4
  for (int i = 0; i < 64; i++) {              // Ks: 16384 u32
    int idx = tid + i * 256;
    int r = idx >> 5, du = idx & 31;
    int grow = blk * 256 - 256 + r;
    Ksw[r * 32 + du] =
        (grow >= 0) ? qkvu[(size_t)grow * 1536 + 512 + h * 32 + du] : 0u;
  }
#pragma unroll 4
  for (int i = 0; i < 128; i++) {             // VT: 32768 halfs (transpose)
    int idx = tid + i * 256;
    int r = idx >> 6, d = idx & 63;
    int grow = blk * 256 - 256 + r;
    VT[d * 514 + r] =
        (grow >= 0) ? qkv[(size_t)grow * 3072 + 2048 + h * 64 + d]
                    : (unsigned short)0;
  }
  __syncthreads();

  // Q fragments live in registers (2 m-tiles x 2 k-steps)
  U16 qa[2][2];
#pragma unroll
  for (int mt = 0; mt < 2; mt++) {
    const size_t qrow = (size_t)(blk * 256 + w * 32 + mt * 16 + ln);
#pragma unroll
    for (int ks = 0; ks < 2; ks++) {
#pragma unroll
      for (int i = 0; i < 4; i++) {
        qa[mt][ks].u[i]     = qkvu[qrow * 1536 + h * 32 + ks * 16 + hi * 4 + i];
        qa[mt][ks].u[4 + i] = qkvu[qrow * 1536 + h * 32 + ks * 16 + 8 + hi * 4 + i];
      }
    }
  }

  v8f O[2][4];
  float runm[2][8], lsum[2][8];
#pragma unroll
  for (int mt = 0; mt < 2; mt++) {
#pragma unroll
    for (int dt = 0; dt < 4; dt++) O[mt][dt] = v8f_zero();
#pragma unroll
    for (int j = 0; j < 8; j++) { runm[mt][j] = -INFINITY; lsum[mt][j] = 0.f; }
  }

  const unsigned* Ksu = (const unsigned*)Ks;
  const unsigned* VTu = (const unsigned*)VT;
  const unsigned* Pbu = (const unsigned*)Pb;
  const float NEG = -INFINITY;

  for (int kt = 0; kt < 16; kt++) {           // 32-key tiles over 512 window
    const int kLo = kt * 32, kHi = kt * 32 + 31;
    if (!(kHi > w * 32 && kLo <= w * 32 + 31 + 256)) continue;  // band skip
    if (blk == 0 && kHi < 256) continue;                        // no prev block

    // ---- S = Q K^T (2 m-tiles x 2 n-tiles, head-dim 64 = 2 WMMA k-steps)
    v8f Sv[2][2];
#pragma unroll
    for (int mt = 0; mt < 2; mt++) {
#pragma unroll
      for (int nt = 0; nt < 2; nt++) {
        U16 b;
        const int key = kt * 32 + nt * 16 + ln;
#pragma unroll
        for (int j = 0; j < 8; j++) b.u[j] = Ksu[key * 32 + (j >> 2) * 8 + hi * 4 + (j & 3)];
        v8f t = wmma_bf16(qa[mt][0].v, b.v, v8f_zero());
#pragma unroll
        for (int j = 0; j < 8; j++) b.u[j] = Ksu[key * 32 + 16 + (j >> 2) * 8 + hi * 4 + (j & 3)];
        Sv[mt][nt] = wmma_bf16(qa[mt][1].v, b.v, t);
      }
    }

    // ---- mask + online softmax (row m = 8*hi+j lives in one 16-lane group)
#pragma unroll
    for (int mt = 0; mt < 2; mt++) {
#pragma unroll
      for (int j = 0; j < 8; j++) {
        const int qi = w * 32 + mt * 16 + hi * 8 + j;
#pragma unroll
        for (int nt = 0; nt < 2; nt++) {
          const int kk = kt * 32 + nt * 16 + ln;
          const bool valid =
              (kk > qi) && (kk <= qi + 256) && (blk > 0 || kk >= 256);
          Sv[mt][nt][j] = valid ? Sv[mt][nt][j] * 0.125f : NEG;
        }
        float m = fmaxf(Sv[mt][0][j], Sv[mt][1][j]);
#pragma unroll
        for (int off = 1; off < 16; off <<= 1) m = fmaxf(m, __shfl_xor(m, off));
        const float newm = fmaxf(runm[mt][j], m);
        const float alpha = (newm == NEG) ? 0.f : __expf(runm[mt][j] - newm);
        float ps = 0.f;
#pragma unroll
        for (int nt = 0; nt < 2; nt++) {
          const float s = Sv[mt][nt][j];
          const float p = (s == NEG) ? 0.f : __expf(s - newm);
          ps += p;
          Pb[w * 1024 + (mt * 16 + hi * 8 + j) * 32 + nt * 16 + ln] = f2bf(p);
        }
#pragma unroll
        for (int off = 1; off < 16; off <<= 1) ps += __shfl_xor(ps, off);
        lsum[mt][j] = lsum[mt][j] * alpha + ps;
        runm[mt][j] = newm;
#pragma unroll
        for (int dt = 0; dt < 4; dt++) O[mt][dt][j] *= alpha;
      }
    }

    // wave-private LDS RAW: LDS is in-order per wave; fence the compiler + cnt
    asm volatile("s_wait_dscnt 0" ::: "memory");

    // ---- O += P V  (P re-read as A fragment, Vt as B fragment)
#pragma unroll
    for (int mt = 0; mt < 2; mt++) {
      U16 pa;
      const int prow = mt * 16 + ln;
#pragma unroll
      for (int i = 0; i < 4; i++) {
        pa.u[i]     = Pbu[w * 512 + prow * 16 + hi * 4 + i];
        pa.u[4 + i] = Pbu[w * 512 + prow * 16 + 8 + hi * 4 + i];
      }
#pragma unroll
      for (int dt = 0; dt < 4; dt++) {
        U16 b;
        const int d = dt * 16 + ln;
#pragma unroll
        for (int j = 0; j < 8; j++) b.u[j] = VTu[d * 257 + kt * 16 + (j >> 2) * 8 + hi * 4 + (j & 3)];
        O[mt][dt] = wmma_bf16(pa.v, b.v, O[mt][dt]);
      }
    }
  }

  // ---- epilogue: divide by row sums, write bf16
#pragma unroll
  for (int mt = 0; mt < 2; mt++) {
#pragma unroll
    for (int j = 0; j < 8; j++) {
      const float lv = lsum[mt][j];
      const float inv = (lv > 0.f) ? 1.f / lv : 0.f;
      const int row = blk * 256 + w * 32 + mt * 16 + hi * 8 + j;
#pragma unroll
      for (int dt = 0; dt < 4; dt++) {
        const int col = h * 64 + dt * 16 + ln;
        o[(size_t)row * 1024 + col] = f2bf(O[mt][dt][j] * inv);
      }
    }
  }
}

// ---------------------------------------------------------------------------
// LayerNorm over 1024 cols, one row per block, bf16 output.
// ---------------------------------------------------------------------------
__global__ __launch_bounds__(256) void layernorm_bf16(
    const float* __restrict__ x, const float* __restrict__ g,
    const float* __restrict__ b, unsigned short* __restrict__ out) {
  __shared__ float red[256];
  const int r = blockIdx.x, tid = threadIdx.x;
  const float* xr = x + (size_t)r * 1024;

  float s = 0.f;
  for (int i = tid; i < 1024; i += 256) s += xr[i];
  red[tid] = s; __syncthreads();
  for (int o = 128; o > 0; o >>= 1) { if (tid < o) red[tid] += red[tid + o]; __syncthreads(); }
  const float mu = red[0] * (1.f / 1024.f);
  __syncthreads();

  float v = 0.f;
  for (int i = tid; i < 1024; i += 256) { float d = xr[i] - mu; v += d * d; }
  red[tid] = v; __syncthreads();
  for (int o = 128; o > 0; o >>= 1) { if (tid < o) red[tid] += red[tid + o]; __syncthreads(); }
  const float rstd = rsqrtf(red[0] * (1.f / 1024.f) + 1e-5f);

  for (int i = tid; i < 1024; i += 256)
    out[(size_t)r * 1024 + i] = f2bf((xr[i] - mu) * rstd * g[i] + b[i]);
}

// ---------------------------------------------------------------------------
// fp32 -> bf16 conversion (weights)
// ---------------------------------------------------------------------------
__global__ void f32_to_bf16(const float* __restrict__ in,
                            unsigned short* __restrict__ out, int n) {
  int i = blockIdx.x * 256 + threadIdx.x;
  if (i < n) out[i] = f2bf(in[i]);
}

// ---------------------------------------------------------------------------
// 3-phase blocked SSM scan:  s_t = A*s_{t-1} + (xn@B_w)_t + u_t
// 32 chunks x 128 steps; serial depth ~130 instead of 4096.
// ---------------------------------------------------------------------------
__global__ void scan_partial(const float* __restrict__ A,
                             const float* __restrict__ drive,
                             const float* __restrict__ ug,
                             float* __restrict__ chunkfin) {
  const int c = blockIdx.x, j = threadIdx.x;
  const float a = A[j];
  float s = 0.f;
  for (int i = 0; i < 128; i++) {
    const int t = c * 128 + i;
    s = a * s + drive[(size_t)t * 128 + j] + ug[(size_t)t * 1152 + j];
  }
  chunkfin[c * 128 + j] = s;
}

__global__ void scan_carry(const float* __restrict__ A,
                           const float* __restrict__ chunkfin,
                           float* __restrict__ carry) {
  const int j = threadIdx.x;
  const float a = A[j];
  float p = a;
  for (int k = 0; k < 7; k++) p = p * p;      // a^128 via 7 squarings
  float cv = 0.f;
  for (int c = 0; c < 32; c++) {
    carry[c * 128 + j] = cv;
    cv = chunkfin[c * 128 + j] + p * cv;
  }
}

__global__ void scan_final(const float* __restrict__ A,
                           const float* __restrict__ drive,
                           const float* __restrict__ ug,
                           const float* __restrict__ carry,
                           unsigned short* __restrict__ states) {
  const int c = blockIdx.x, j = threadIdx.x;
  const float a = A[j];
  float s = carry[c * 128 + j];
  for (int i = 0; i < 128; i++) {
    const int t = c * 128 + i;
    s = a * s + drive[(size_t)t * 128 + j] + ug[(size_t)t * 1152 + j];
    states[(size_t)t * 128 + j] = f2bf(s);
  }
}

// ---------------------------------------------------------------------------
// xmid = x + g*attn + (1-g)*y_ssm,  g = sigmoid(ug[:,128:])
// ---------------------------------------------------------------------------
__global__ void combine_mix(const float* __restrict__ x,
                            const float* __restrict__ attn,
                            const float* __restrict__ yssm,
                            const float* __restrict__ ugf,
                            float* __restrict__ xmid) {
  const int i = blockIdx.x * 256 + threadIdx.x;   // < 4096*1024
  const int t = i >> 10, c = i & 1023;
  const float g = 1.f / (1.f + __expf(-ugf[(size_t)t * 1152 + 128 + c]));
  xmid[i] = x[i] + g * attn[i] + (1.f - g) * yssm[i];
}

// ---------------------------------------------------------------------------
// Host launcher
// ---------------------------------------------------------------------------
extern "C" void kernel_launch(void* const* d_in, const int* in_sizes, int n_in,
                              void* d_out, int out_size, void* d_ws,
                              size_t ws_size, hipStream_t stream) {
  (void)in_sizes; (void)n_in; (void)out_size; (void)ws_size;

  const float* x      = (const float*)d_in[0];
  const float* ln1_g  = (const float*)d_in[1];
  const float* ln1_b  = (const float*)d_in[2];
  const float* ln2_g  = (const float*)d_in[3];
  const float* ln2_b  = (const float*)d_in[4];
  const float* W_qkv  = (const float*)d_in[5];
  const float* W_O    = (const float*)d_in[6];
  const float* b_O    = (const float*)d_in[7];
  const float* W_ug   = (const float*)d_in[8];
  const float* b_ug   = (const float*)d_in[9];
  const float* B_w    = (const float*)d_in[10];
  const float* Adiag  = (const float*)d_in[11];
  const float* C_w    = (const float*)d_in[12];
  const float* W1     = (const float*)d_in[13];
  const float* b1     = (const float*)d_in[14];
  const float* W2     = (const float*)d_in[15];
  const float* b2     = (const float*)d_in[16];
  float* out = (float*)d_out;

  char* p = (char*)d_ws;
  auto take = [&](size_t bytes) -> void* {
    void* r = (void*)p;
    p += (bytes + 255) & ~(size_t)255;
    return r;
  };

  unsigned short* xn_b     = (unsigned short*)take((size_t)4096 * 1024 * 2);
  unsigned short* wqkv_b   = (unsigned short*)take((size_t)1024 * 3072 * 2);
  unsigned short* wo_b     = (unsigned short*)take((size_t)1024 * 1024 * 2);
  unsigned short* wug_b    = (unsigned short*)take((size_t)1024 * 1152 * 2);
  unsigned short* bw_b     = (unsigned short*)take((size_t)1024 * 128 * 2);
  unsigned short* cw_b     = (unsigned short*)take((size_t)128 * 1024 * 2);
  unsigned short* w1_b     = (unsigned short*)take((size_t)1024 * 4096 * 2);
  unsigned short* w2_b     = (unsigned short*)take((size_t)4096 * 1024 * 2);
  unsigned short* qkv_b    = (unsigned short*)take((size_t)4096 * 3072 * 2);
  unsigned short* o_b      = (unsigned short*)take((size_t)4096 * 1024 * 2);
  float*          attn_f   = (float*)take((size_t)4096 * 1024 * 4);
  float*          ug_f     = (float*)take((size_t)4096 * 1152 * 4);
  float*          drive_f  = (float*)take((size_t)4096 * 128 * 4);
  float*          chunkfin = (float*)take((size_t)32 * 128 * 4);
  float*          carry    = (float*)take((size_t)32 * 128 * 4);
  unsigned short* states_b = (unsigned short*)take((size_t)4096 * 128 * 2);
  float*          yssm_f   = (float*)take((size_t)4096 * 1024 * 4);
  float*          xmid_f   = (float*)take((size_t)4096 * 1024 * 4);
  unsigned short* h_b      = (unsigned short*)take((size_t)4096 * 1024 * 2);
  unsigned short* hid_b    = (unsigned short*)take((size_t)4096 * 4096 * 2);

  auto cvt = [&](const float* src, unsigned short* dst, int n) {
    f32_to_bf16<<<(n + 255) / 256, 256, 0, stream>>>(src, dst, n);
  };
  cvt(W_qkv, wqkv_b, 1024 * 3072);
  cvt(W_O,   wo_b,   1024 * 1024);
  cvt(W_ug,  wug_b,  1024 * 1152);
  cvt(B_w,   bw_b,   1024 * 128);
  cvt(C_w,   cw_b,   128 * 1024);
  cvt(W1,    w1_b,   1024 * 4096);
  cvt(W2,    w2_b,   4096 * 1024);

  // LN1 -> xn (bf16)
  layernorm_bf16<<<4096, 256, 0, stream>>>(x, ln1_g, ln1_b, xn_b);

  // qkv = xn @ W_qkv            (bf16 out for attention)
  gemm_bf16_wmma<<<dim3(3072 / TBN, 4096 / TBM), 256, 0, stream>>>(
      xn_b, wqkv_b, nullptr, nullptr, nullptr, qkv_b, 4096, 3072, 1024, 0);

  // windowed attention -> o (bf16)
  attn_wmma<<<dim3(16, 16), 256, 0, stream>>>(qkv_b, o_b);

  // attn = o @ W_O + b_O        (f32)
  gemm_bf16_wmma<<<dim3(1024 / TBN, 4096 / TBM), 256, 0, stream>>>(
      o_b, wo_b, b_O, nullptr, attn_f, nullptr, 4096, 1024, 1024, 0);

  // ug = xn @ W_ug + b_ug       (f32: cols 0..127 = u, 128.. = pre-sigmoid g)
  gemm_bf16_wmma<<<dim3(1152 / TBN, 4096 / TBM), 256, 0, stream>>>(
      xn_b, wug_b, b_ug, nullptr, ug_f, nullptr, 4096, 1152, 1024, 0);

  // drive = xn @ B_w            (f32; u added inside scan)
  gemm_bf16_wmma<<<dim3(128 / TBN, 4096 / TBM), 256, 0, stream>>>(
      xn_b, bw_b, nullptr, nullptr, drive_f, nullptr, 4096, 128, 1024, 0);

  // blocked SSM scan -> states (bf16)
  scan_partial<<<32, 128, 0, stream>>>(Adiag, drive_f, ug_f, chunkfin);
  scan_carry<<<1, 128, 0, stream>>>(Adiag, chunkfin, carry);
  scan_final<<<32, 128, 0, stream>>>(Adiag, drive_f, ug_f, carry, states_b);

  // y_ssm = states @ C_w        (f32)
  gemm_bf16_wmma<<<dim3(1024 / TBN, 4096 / TBM), 256, 0, stream>>>(
      states_b, cw_b, nullptr, nullptr, yssm_f, nullptr, 4096, 1024, 128, 0);

  // xmid = x + g*attn + (1-g)*y_ssm
  combine_mix<<<(4096 * 1024) / 256, 256, 0, stream>>>(x, attn_f, yssm_f, ug_f,
                                                       xmid_f);

  // LN2 -> h (bf16)
  layernorm_bf16<<<4096, 256, 0, stream>>>(xmid_f, ln2_g, ln2_b, h_b);

  // hidden = gelu(h @ W1 + b1)  (bf16)
  gemm_bf16_wmma<<<dim3(4096 / TBN, 4096 / TBM), 256, 0, stream>>>(
      h_b, w1_b, b1, nullptr, nullptr, hid_b, 4096, 4096, 1024, 1);

  // out = xmid + hidden @ W2 + b2   (f32, final output)
  gemm_bf16_wmma<<<dim3(1024 / TBN, 4096 / TBM), 256, 0, stream>>>(
      hid_b, w2_b, b2, xmid_f, out, nullptr, 4096, 1024, 4096, 0);
}
